// HOIMamba_homo_36352603193655
// MI455X (gfx1250) — compile-verified, matrix operand order checked by reading, not run on hardware
//
#include <hip/hip_runtime.h>
#include <hip/hip_bf16.h>
#include <math.h>

// ---------------- problem constants ----------------
#define B_SZ   4
#define T_LEN  196
#define L_SEQ  392            // TM + T
#define DM     1024           // D_MODEL
#define DI     2048           // D_INNER
#define NXZ    4096           // 2*D_INNER (in_proj output)
#define DSTATE 16
#define RANK   64             // DT_RANK
#define NPROJ  96             // RANK + 2*DSTATE
#define MROWS  (B_SZ*L_SEQ)   // 1568 (multiple of 16)

typedef unsigned short ushort_t;                      // bf16 storage
typedef __bf16 v8bf  __attribute__((ext_vector_type(8)));
typedef __bf16 v16bf __attribute__((ext_vector_type(16)));
typedef float  v8f   __attribute__((ext_vector_type(8)));

__device__ __forceinline__ ushort_t f2bf(float f) {   // RNE f32 -> bf16
  union { float f; unsigned u; } x; x.f = f;
  unsigned r = x.u + 0x7FFFu + ((x.u >> 16) & 1u);
  return (ushort_t)(r >> 16);
}
__device__ __forceinline__ float silu(float x) { return x / (1.f + __expf(-x)); }

// block-wide sum over 256 threads (8 waves), result broadcast to all threads
__device__ __forceinline__ float block_sum(float v, float* s8) {
  const int lane = threadIdx.x & 31, w = threadIdx.x >> 5;
  #pragma unroll
  for (int o = 16; o; o >>= 1) v += __shfl_xor(v, o, 32);
  __syncthreads();
  if (lane == 0) s8[w] = v;
  __syncthreads();
  float r = s8[0];
  #pragma unroll
  for (int i = 1; i < 8; ++i) r += s8[i];
  return r;
}

// ---------------- tiny elementwise kernels ----------------

// timestep embedding: temb[b][0:512]=cos, [512:1024]=sin
__global__ __launch_bounds__(512) void k_temb(const int* __restrict__ ts,
                                              float* __restrict__ temb) {
  const int b = blockIdx.x, i = threadIdx.x;          // i in [0,512)
  const float t = (float)ts[b];
  const float fr = __expf(-logf(10000.f) * (float)i / 512.f);
  const float a = t * fr;
  temb[b * 1024 + i]       = cosf(a);
  temb[b * 1024 + 512 + i] = sinf(a);
}

// one wave per output row-dot: out[b][o] = act(dot(X[b,:],W[o,:]) + bias[o])
template<bool SILU>
__global__ __launch_bounds__(256) void k_rowdot(const float* __restrict__ X,
                                                const float* __restrict__ W,
                                                const float* __restrict__ bias,
                                                float* __restrict__ out,
                                                int Bn, int N, int K) {
  const int wid = blockIdx.x * 8 + (threadIdx.x >> 5);
  const int lane = threadIdx.x & 31;
  if (wid >= Bn * N) return;
  const int b = wid / N, o = wid % N;
  const float* x = X + (size_t)b * K;
  const float* w = W + (size_t)o * K;
  float s = 0.f;
  for (int k = lane; k < K; k += 32) s += x[k] * w[k];
  #pragma unroll
  for (int off = 16; off; off >>= 1) s += __shfl_xor(s, off, 32);
  if (lane == 0) {
    s += bias[o];
    if (SILU) s = silu(s);
    out[wid] = s;
  }
}

// build seq row (pos_emb + input [+emb_t]) -> LayerNorm -> seq(f32)
// then RMSNorm of that row -> seq_bf (bf16, GEMM operand)
__global__ __launch_bounds__(256) void k_build_seq(
    const float* __restrict__ x_r, const float* __restrict__ motion,
    const float* __restrict__ pos, const float* __restrict__ emb_t,
    const float* __restrict__ g, const float* __restrict__ bta,
    const float* __restrict__ rms_w,
    float* __restrict__ seq, ushort_t* __restrict__ seq_bf) {
  __shared__ float s8[8];
  const int row = blockIdx.x;                // b*392 + l
  const int b = row / L_SEQ, l = row % L_SEQ;
  float v[4];
  #pragma unroll
  for (int j = 0; j < 4; ++j) {
    const int d = threadIdx.x + j * 256;
    float val;
    if (l < T_LEN) {
      val = pos[l * DM + d] + motion[((size_t)b * T_LEN + l) * DM + d];
    } else {
      const int lr = l - T_LEN;
      val = pos[lr * DM + d] + x_r[((size_t)b * T_LEN + lr) * DM + d] + emb_t[b * DM + d];
    }
    v[j] = val;
  }
  float s = v[0] + v[1] + v[2] + v[3];
  float s2 = v[0]*v[0] + v[1]*v[1] + v[2]*v[2] + v[3]*v[3];
  const float S  = block_sum(s,  s8);
  const float S2 = block_sum(s2, s8);
  const float mu = S * (1.f / DM);
  const float var = S2 * (1.f / DM) - mu * mu;
  const float inv = rsqrtf(var + 1e-5f);
  float ln[4], q = 0.f;
  #pragma unroll
  for (int j = 0; j < 4; ++j) {
    const int d = threadIdx.x + j * 256;
    ln[j] = (v[j] - mu) * inv * g[d] + bta[d];
    q += ln[j] * ln[j];
  }
  const float Q = block_sum(q, s8);
  const float rinv = rsqrtf(Q * (1.f / DM) + 1e-5f);
  #pragma unroll
  for (int j = 0; j < 4; ++j) {
    const int d = threadIdx.x + j * 256;
    seq[(size_t)row * DM + d] = ln[j];
    seq_bf[(size_t)row * DM + d] = f2bf(ln[j] * rinv * rms_w[d]);
  }
}

// flat f32 -> bf16
__global__ __launch_bounds__(256) void k_cvt_bf16(const float* __restrict__ in,
                                                  ushort_t* __restrict__ out, int n) {
  const int i = blockIdx.x * 256 + threadIdx.x;
  if (i < n) out[i] = f2bf(in[i]);
}

// dt slice of dbc -> bf16 (columns 0..63 of the 96-wide dbc)
__global__ __launch_bounds__(256) void k_extract_dt(const float* __restrict__ dbc,
                                                    ushort_t* __restrict__ dt) {
  const int i = blockIdx.x * 256 + threadIdx.x;     // over MROWS*RANK
  if (i < MROWS * RANK) {
    const int m = i >> 6, r = i & 63;
    dt[i] = f2bf(dbc[(size_t)m * NPROJ + r]);
  }
}

// causal depthwise conv (D_CONV=4, within each batch's 392 seq) + bias + SiLU
__global__ __launch_bounds__(256) void k_conv_silu(const float* __restrict__ xz,
                                                   const float* __restrict__ cw,
                                                   const float* __restrict__ cb,
                                                   float* __restrict__ xc,
                                                   ushort_t* __restrict__ xc_bf) {
  const size_t i = (size_t)blockIdx.x * 256 + threadIdx.x;   // MROWS*DI
  if (i >= (size_t)MROWS * DI) return;
  const int e = (int)(i & (DI - 1));
  const int row = (int)(i >> 11);
  const int l = row % L_SEQ, b = row / L_SEQ;
  float acc = cb[e];
  #pragma unroll
  for (int j = 0; j < 4; ++j) {
    const int ls = l - 3 + j;
    if (ls >= 0)
      acc += cw[e * 4 + j] * xz[((size_t)(b * L_SEQ + ls)) * NXZ + e];
  }
  const float sv = silu(acc);
  xc[i] = sv;
  xc_bf[i] = f2bf(sv);
}

// selective scan: one thread per (b,e) channel; 16 fp32 states in VGPRs,
// per-step B/C (32 floats) staged through LDS by lanes 0..31.
__global__ __launch_bounds__(256) void k_scan(const float* __restrict__ delta,
                                              const float* __restrict__ xc,
                                              const float* __restrict__ xz,
                                              const float* __restrict__ dbc,
                                              const float* __restrict__ A_log,
                                              const float* __restrict__ Dp,
                                              ushort_t* __restrict__ y_bf) {
  const int b = blockIdx.y;
  const int e = blockIdx.x * 256 + threadIdx.x;     // < DI
  float An[DSTATE];
  #pragma unroll
  for (int n = 0; n < DSTATE; ++n) An[n] = -__expf(A_log[e * DSTATE + n]);
  const float Dv = Dp[e];
  float h[DSTATE];
  #pragma unroll
  for (int n = 0; n < DSTATE; ++n) h[n] = 0.f;
  __shared__ float BC[32];
  for (int l = 0; l < L_SEQ; ++l) {
    const size_t row = (size_t)b * L_SEQ + l;
    if (threadIdx.x < 32) BC[threadIdx.x] = dbc[row * NPROJ + RANK + threadIdx.x];
    __syncthreads();
    const float d = delta[row * DI + e];
    const float u = xc[row * DI + e];
    const float zv = xz[row * NXZ + DI + e];
    float y = 0.f;
    #pragma unroll
    for (int n = 0; n < DSTATE; ++n) {
      const float dA = __expf(d * An[n]);
      h[n] = dA * h[n] + (d * BC[n]) * u;
      y += h[n] * BC[16 + n];
    }
    y = (y + u * Dv) * silu(zv);
    y_bf[row * DI + e] = f2bf(y);
    __syncthreads();
  }
}

// final LayerNorm over seq_out rows [b*392+196+l] -> out[b][l][:]
__global__ __launch_bounds__(256) void k_final_ln(const float* __restrict__ seq_out,
                                                  const float* __restrict__ g,
                                                  const float* __restrict__ bta,
                                                  float* __restrict__ out) {
  __shared__ float s8[8];
  const int b = blockIdx.x / T_LEN, l = blockIdx.x % T_LEN;
  const float* src = seq_out + ((size_t)b * L_SEQ + T_LEN + l) * DM;
  float v[4];
  #pragma unroll
  for (int j = 0; j < 4; ++j) v[j] = src[threadIdx.x + j * 256];
  float s = v[0] + v[1] + v[2] + v[3];
  float s2 = v[0]*v[0] + v[1]*v[1] + v[2]*v[2] + v[3]*v[3];
  const float S  = block_sum(s,  s8);
  const float S2 = block_sum(s2, s8);
  const float mu = S * (1.f / DM);
  const float inv = rsqrtf(S2 * (1.f / DM) - mu * mu + 1e-5f);
  float* dst = out + ((size_t)b * T_LEN + l) * DM;
  #pragma unroll
  for (int j = 0; j < 4; ++j) {
    const int d = threadIdx.x + j * 256;
    dst[d] = (v[j] - mu) * inv * g[d] + bta[d];
  }
}

// ---------------- WMMA bf16 GEMM: D(MxN) = A(MxK) * Bw(NxK)^T [+epilogue] ----
// Per ISA 7.12.2 (16-bit A 16x32): lanes 0-15 hold K {k..k+7, k+16..k+23},
// lanes 16-31 hold K {k+8..k+15, k+24..k+31}; B (weights stored N-major,
// K-contiguous) uses the identical per-lane pattern with N = lane.
// NSUB = number of 16-col subtiles per wave (compile-time -> branch-free k-loop).
// Inner loop processes K in 64-deep ping-pong steps: ALL fragment loads of the
// step are issued ahead of the WMMAs (distinct registers) so the first 4 WMMAs
// only need a partial s_wait_loadcnt and the next iteration's loads issue while
// the matrix pipe drains.
enum { EPI_NONE = 0, EPI_BIAS_SOFTPLUS = 1, EPI_RESIDUAL = 2 };

__device__ __forceinline__ v16bf load_frag16(const ushort_t* __restrict__ row, int k) {
  const v8bf c0 = *reinterpret_cast<const v8bf*>(row + k);        // 16B chunk
  const v8bf c1 = *reinterpret_cast<const v8bf*>(row + k + 16);   // 16B chunk
  return __builtin_shufflevector(c0, c1, 0,1,2,3,4,5,6,7,8,9,10,11,12,13,14,15);
}

template<int EPI, int NSUB>
__global__ __launch_bounds__(256)
void k_gemm_bf16(const ushort_t* __restrict__ A, const ushort_t* __restrict__ Bw,
                 float* __restrict__ D, const float* __restrict__ aux,
                 int M, int N, int K) {
  const int lane = threadIdx.x & 31;
  const int wave = threadIdx.x >> 5;
  const int m0 = blockIdx.x * 128 + wave * 16;   // 8 waves -> 128-row block tile
  if (m0 >= M) return;                           // wave-uniform; EXEC stays full
  const int n0 = blockIdx.y * (NSUB * 16);       // NSUB x 16-col subtiles (exact)
  const int half = lane >> 4;
  const int l16 = lane & 15;
  const ushort_t* arow = A + (size_t)(m0 + l16) * K + half * 8;
  const ushort_t* brow[NSUB];
  #pragma unroll
  for (int s = 0; s < NSUB; ++s)
    brow[s] = Bw + (size_t)(n0 + s * 16 + l16) * K + half * 8;

  v8f acc[NSUB];
  #pragma unroll
  for (int s = 0; s < NSUB; ++s) acc[s] = 0.f;

  // K is a multiple of 64 for every launch in this pipeline.
  for (int k = 0; k < K; k += 64) {
    // ---- issue ALL loads of this 64-deep step first (distinct regs) ----
    const v16bf af0 = load_frag16(arow, k);
    v16bf bf0[NSUB];
    #pragma unroll
    for (int s = 0; s < NSUB; ++s) bf0[s] = load_frag16(brow[s], k);
    const v16bf af1 = load_frag16(arow, k + 32);
    v16bf bf1[NSUB];
    #pragma unroll
    for (int s = 0; s < NSUB; ++s) bf1[s] = load_frag16(brow[s], k + 32);
    // ---- consume: first half then second half ----
    #pragma unroll
    for (int s = 0; s < NSUB; ++s)
      acc[s] = __builtin_amdgcn_wmma_f32_16x16x32_bf16(
          false, af0, false, bf0[s], (short)0, acc[s], false, false);
    #pragma unroll
    for (int s = 0; s < NSUB; ++s)
      acc[s] = __builtin_amdgcn_wmma_f32_16x16x32_bf16(
          false, af1, false, bf1[s], (short)0, acc[s], false, false);
  }
  // C/D layout: VGPR r, lanes 0-15 -> (M=m0+r, N=l16); lanes 16-31 -> M=m0+8+r
  const int mrb = m0 + (half ? 8 : 0);
  #pragma unroll
  for (int s = 0; s < NSUB; ++s) {
    const int n = n0 + s * 16 + l16;
    #pragma unroll
    for (int r = 0; r < 8; ++r) {
      const int m = mrb + r;
      float v = acc[s][r];
      if (EPI == EPI_BIAS_SOFTPLUS) {
        v += aux[n];
        v = (v > 20.f) ? v : log1pf(__expf(v));
      } else if (EPI == EPI_RESIDUAL) {
        v += aux[(size_t)m * N + n];
      }
      D[(size_t)m * N + n] = v;
    }
  }
}

// ---------------- launcher ----------------
extern "C" void kernel_launch(void* const* d_in, const int* in_sizes, int n_in,
                              void* d_out, int out_size, void* d_ws, size_t ws_size,
                              hipStream_t stream) {
  const float* x_r    = (const float*)d_in[0];
  const int*   tsteps = (const int*)  d_in[1];
  const float* motion = (const float*)d_in[2];
  const float* tw1    = (const float*)d_in[3];
  const float* tb1    = (const float*)d_in[4];
  const float* tw2    = (const float*)d_in[5];
  const float* tb2    = (const float*)d_in[6];
  const float* pos    = (const float*)d_in[7];
  const float* ln_g   = (const float*)d_in[8];
  const float* ln_b   = (const float*)d_in[9];
  const float* w_in   = (const float*)d_in[10];
  const float* conv_w = (const float*)d_in[11];
  const float* conv_b = (const float*)d_in[12];
  const float* w_xp   = (const float*)d_in[13];
  const float* w_dt   = (const float*)d_in[14];
  const float* dt_b   = (const float*)d_in[15];
  const float* A_log  = (const float*)d_in[16];
  const float* Dp     = (const float*)d_in[17];
  const float* w_out  = (const float*)d_in[18];
  const float* rms_w  = (const float*)d_in[19];
  float* out = (float*)d_out;

  char* p = (char*)d_ws;
  auto carve = [&](size_t bytes) -> void* {
    void* r = (void*)p;
    p += (bytes + 255) & ~(size_t)255;
    return r;
  };
  float*    temb    = (float*)   carve((size_t)B_SZ * 1024 * 4);
  float*    h_time  = (float*)   carve((size_t)B_SZ * 2048 * 4);
  float*    emb_t   = (float*)   carve((size_t)B_SZ * 1024 * 4);
  float*    seq     = (float*)   carve((size_t)MROWS * DM * 4);
  ushort_t* seq_bf  = (ushort_t*)carve((size_t)MROWS * DM * 2);
  ushort_t* w_in_bf = (ushort_t*)carve((size_t)NXZ * DM * 2);
  ushort_t* w_xp_bf = (ushort_t*)carve((size_t)NPROJ * DI * 2);
  ushort_t* w_dt_bf = (ushort_t*)carve((size_t)DI * RANK * 2);
  ushort_t* w_out_bf= (ushort_t*)carve((size_t)DM * DI * 2);
  float*    xz      = (float*)   carve((size_t)MROWS * NXZ * 4);
  float*    xc      = (float*)   carve((size_t)MROWS * DI * 4);
  ushort_t* xc_bf   = (ushort_t*)carve((size_t)MROWS * DI * 2);
  float*    dbc     = (float*)   carve((size_t)MROWS * NPROJ * 4);
  ushort_t* dt_bf   = (ushort_t*)carve((size_t)MROWS * RANK * 2);
  float*    delta   = (float*)   carve((size_t)MROWS * DI * 4);
  ushort_t* y_bf    = (ushort_t*)carve((size_t)MROWS * DI * 2);
  float*    seq_out = (float*)   carve((size_t)MROWS * DM * 4);
  (void)ws_size; (void)in_sizes; (void)n_in; (void)out_size;

  // 1) timestep embedding + time MLP (tiny, fp32 wave-dots)
  k_temb<<<B_SZ, 512, 0, stream>>>(tsteps, temb);
  k_rowdot<true ><<<(B_SZ * 2048) / 8, 256, 0, stream>>>(temb,   tw1, tb1, h_time, B_SZ, 2048, 1024);
  k_rowdot<false><<<(B_SZ * 1024) / 8, 256, 0, stream>>>(h_time, tw2, tb2, emb_t,  B_SZ, 1024, 2048);

  // 2) build seq + LayerNorm + RMSNorm (bf16 GEMM operand)
  k_build_seq<<<MROWS, 256, 0, stream>>>(x_r, motion, pos, emb_t, ln_g, ln_b, rms_w,
                                         seq, seq_bf);

  // 3) weight conversions f32 -> bf16
  k_cvt_bf16<<<(NXZ * DM + 255) / 256,   256, 0, stream>>>(w_in,  w_in_bf,  NXZ * DM);
  k_cvt_bf16<<<(NPROJ * DI + 255) / 256, 256, 0, stream>>>(w_xp,  w_xp_bf,  NPROJ * DI);
  k_cvt_bf16<<<(DI * RANK + 255) / 256,  256, 0, stream>>>(w_dt,  w_dt_bf,  DI * RANK);
  k_cvt_bf16<<<(DM * DI + 255) / 256,    256, 0, stream>>>(w_out, w_out_bf, DM * DI);

  const dim3 blk(256);
  // 4) in_proj: (1568x1024) x (4096x1024)^T -> xz          [NSUB=4 -> 64 cols]
  k_gemm_bf16<EPI_NONE, 4><<<dim3((MROWS + 127) / 128, NXZ / 64), blk, 0, stream>>>(
      seq_bf, w_in_bf, xz, nullptr, MROWS, NXZ, DM);

  // 5) causal depthwise conv + SiLU
  k_conv_silu<<<(MROWS * DI) / 256, 256, 0, stream>>>(xz, conv_w, conv_b, xc, xc_bf);

  // 6) x_proj: (1568x2048) x (96x2048)^T -> dbc            [NSUB=2 -> 32 cols x 3]
  k_gemm_bf16<EPI_NONE, 2><<<dim3((MROWS + 127) / 128, NPROJ / 32), blk, 0, stream>>>(
      xc_bf, w_xp_bf, dbc, nullptr, MROWS, NPROJ, DI);

  // 7) dt slice -> bf16; dt_proj + bias + softplus -> delta
  k_extract_dt<<<(MROWS * RANK) / 256, 256, 0, stream>>>(dbc, dt_bf);
  k_gemm_bf16<EPI_BIAS_SOFTPLUS, 4><<<dim3((MROWS + 127) / 128, DI / 64), blk, 0, stream>>>(
      dt_bf, w_dt_bf, delta, dt_b, MROWS, DI, RANK);

  // 8) selective scan (+ D skip + SiLU(z) gate), bf16 output for out_proj
  k_scan<<<dim3(DI / 256, B_SZ), 256, 0, stream>>>(delta, xc, xz, dbc, A_log, Dp, y_bf);

  // 9) out_proj + residual: seq_out = seq + y @ out_proj_w^T
  k_gemm_bf16<EPI_RESIDUAL, 4><<<dim3((MROWS + 127) / 128, DM / 64), blk, 0, stream>>>(
      y_bf, w_out_bf, seq_out, seq, MROWS, DM, DI);

  // 10) final LayerNorm on the regular-token half
  k_final_ln<<<B_SZ * T_LEN, 256, 0, stream>>>(seq_out, ln_g, ln_b, out);
}
